// UHGTransformer_65438121721902
// MI455X (gfx1250) — matrix operand order repaired and checked by reading, not compiled, not run on hardware
//
#include <hip/hip_runtime.h>
#include <hip/hip_bf16.h>
#include <math.h>

typedef __attribute__((ext_vector_type(16))) __bf16 v16bf;
typedef __attribute__((ext_vector_type(8)))  __bf16 v8bf;
typedef __attribute__((ext_vector_type(8)))  float  v8f;
typedef __attribute__((ext_vector_type(4)))  int    v4i;

#define S_LEN 2048
#define BATCH 2
#define DM    1024
#define NH    16
#define HD    64
#define DFF   4096
#define MROWS (S_LEN * BATCH)   // 4096

// GEMM blocking
#define BM  64
#define BN  128
#define BK  64
#define BKP 72   // padded row stride (halves) to avoid LDS bank conflicts

#define AS1 __attribute__((address_space(1)))
#define AS3 __attribute__((address_space(3)))

#if __has_builtin(__builtin_amdgcn_global_load_async_to_lds_b128)
#define HAS_ASYNC_LDS 1
#else
#define HAS_ASYNC_LDS 0
#endif

#define STR2(x) #x
#define STR(x) STR2(x)
#if HAS_ASYNC_LDS
#if __has_builtin(__builtin_amdgcn_s_wait_asynccnt)
#define WAIT_ASYNC(n) __builtin_amdgcn_s_wait_asynccnt(n)
#else
#define WAIT_ASYNC(n) asm volatile("s_wait_asynccnt " STR(n) ::: "memory")
#endif
#else
#define WAIT_ASYNC(n)
#endif

__device__ __forceinline__ __bf16 to_bf16(float f) { return (__bf16)f; }

__device__ __forceinline__ v8f wmma_bf16(v16bf a, v16bf b, v8f c) {
    return __builtin_amdgcn_wmma_f32_16x16x32_bf16(false, a, false, b, (short)0, c, false, false);
}

// 16-byte global -> LDS copy (async on gfx1250 toolchains that expose it)
__device__ __forceinline__ void copy16_g2l(const __bf16* g, __bf16* l) {
#if HAS_ASYNC_LDS
    __builtin_amdgcn_global_load_async_to_lds_b128((AS1 v4i*)g, (AS3 v4i*)l, 0, 0);
#else
    *(v8bf*)l = *(const v8bf*)g;
#endif
}

// A fragment (16x32 bf16) from LDS tile with padded stride
__device__ __forceinline__ v16bf lds_a_frag(const __bf16* T, int mrow, int kbase, int lane) {
    const __bf16* p = T + (size_t)(mrow + (lane & 15)) * BKP + kbase + ((lane & 16) ? 16 : 0);
    v16bf r;
    #pragma unroll
    for (int t = 0; t < 8; ++t) { r[t] = p[t]; r[8 + t] = p[8 + t]; }
    return r;
}

// B fragment (32x16) from LDS tile holding W rows (N x K layout)
__device__ __forceinline__ v16bf lds_b_frag(const __bf16* T, int nrow, int kbase, int lane) {
    const __bf16* p = T + (size_t)(nrow + (lane & 15)) * BKP + kbase + ((lane & 16) ? 16 : 0);
    v16bf r;
    #pragma unroll
    for (int u = 0; u < 16; ++u) r[u] = p[u];
    return r;
}

// ---------------- elementwise f32 -> bf16 ----------------
__global__ __launch_bounds__(256) void cvt_kernel(const float* __restrict__ in,
                                                  __bf16* __restrict__ out, size_t n) {
    size_t i = (size_t)blockIdx.x * blockDim.x + threadIdx.x;
    if (i < n) out[i] = to_bf16(in[i]);
}

// ---------------- blocked GEMM: C[M,N] = A[M,K] @ W[N,K]^T + bias ----------------
// 8 waves; block tile 64x128; wave tile 32x32; LDS double-buffered K-stages of 64.
__global__ __launch_bounds__(256) void gemm_bias_kernel(const __bf16* __restrict__ A,
                                                        const __bf16* __restrict__ W,
                                                        const float*  __restrict__ bias,
                                                        float* __restrict__ C,
                                                        int M, int N, int K) {
    __shared__ __bf16 As[2][BM * BKP];   // 18 KB
    __shared__ __bf16 Bs[2][BN * BKP];   // 36 KB
    int tid  = threadIdx.x;
    int lane = tid & 31;
    int wave = tid >> 5;
    int wm = wave >> 2;       // 0..1  (M sub-tile)
    int wn = wave & 3;        // 0..3  (N sub-tile)
    int m0 = blockIdx.y * BM;
    int n0 = blockIdx.x * BN;

    auto stage = [&](int buf, int k0) {
        // A tile: 64 rows x 64 halves = 512 x 16B chunks, 2 per thread
        #pragma unroll
        for (int c = 0; c < 2; ++c) {
            int ch = tid + c * 256;
            int row = ch >> 3, col = ch & 7;
            copy16_g2l(A + (size_t)(m0 + row) * K + k0 + col * 8,
                       &As[buf][(size_t)row * BKP + col * 8]);
        }
        // B tile: 128 rows x 64 halves = 1024 chunks, 4 per thread
        #pragma unroll
        for (int c = 0; c < 4; ++c) {
            int ch = tid + c * 256;
            int row = ch >> 3, col = ch & 7;
            copy16_g2l(W + (size_t)(n0 + row) * K + k0 + col * 8,
                       &Bs[buf][(size_t)row * BKP + col * 8]);
        }
    };

    v8f acc00 = {}, acc01 = {}, acc10 = {}, acc11 = {};
    int nstages = K / BK;

    stage(0, 0);
    for (int s = 0; s < nstages; ++s) {
        int buf = s & 1;
        if (s + 1 < nstages) {
            stage(buf ^ 1, (s + 1) * BK);
            WAIT_ASYNC(6);          // our 6 in-flight are next stage's; current is done
        } else {
            WAIT_ASYNC(0);
        }
        __syncthreads();
        #pragma unroll
        for (int kk = 0; kk < 2; ++kk) {
            int kbase = kk * 32;
            v16bf a0 = lds_a_frag(As[buf], wm * 32 + 0,  kbase, lane);
            v16bf a1 = lds_a_frag(As[buf], wm * 32 + 16, kbase, lane);
            v16bf b0 = lds_b_frag(Bs[buf], wn * 32 + 0,  kbase, lane);
            v16bf b1 = lds_b_frag(Bs[buf], wn * 32 + 16, kbase, lane);
            acc00 = wmma_bf16(a0, b0, acc00);
            acc01 = wmma_bf16(a0, b1, acc01);
            acc10 = wmma_bf16(a1, b0, acc10);
            acc11 = wmma_bf16(a1, b1, acc11);
        }
        __syncthreads();            // protect buf before it is overwritten next iter
    }

    int cn = lane & 15, mofs = (lane & 16) ? 8 : 0;
    #pragma unroll
    for (int i = 0; i < 2; ++i) {
        #pragma unroll
        for (int j = 0; j < 2; ++j) {
            const v8f* accp = (i == 0) ? ((j == 0) ? &acc00 : &acc01)
                                       : ((j == 0) ? &acc10 : &acc11);
            int n = n0 + wn * 32 + j * 16 + cn;
            float bvv = bias ? bias[n] : 0.0f;
            #pragma unroll
            for (int r = 0; r < 8; ++r)
                C[(size_t)(m0 + wm * 32 + i * 16 + mofs + r) * N + n] = (*accp)[r] + bvv;
        }
    }
}

// ---------------- row L2-normalize (project to sphere), optional relu/bf16 ----------------
__global__ __launch_bounds__(256) void rownorm_kernel(const float* __restrict__ in,
                                                      float* __restrict__ of,
                                                      __bf16* __restrict__ ob,
                                                      int D, float eps, int relu) {
    size_t row = blockIdx.x;
    const float* x = in + row * (size_t)D;
    float s = 0.0f;
    for (int i = threadIdx.x; i < D; i += 256) { float v = x[i]; s += v * v; }
    #pragma unroll
    for (int o = 16; o; o >>= 1) s += __shfl_xor(s, o, 32);
    __shared__ float red[8];
    __shared__ float bc;
    if ((threadIdx.x & 31) == 0) red[threadIdx.x >> 5] = s;
    __syncthreads();
    if (threadIdx.x == 0) {
        float t = 0.0f;
        #pragma unroll
        for (int i = 0; i < 8; ++i) t += red[i];
        bc = 1.0f / (sqrtf(t) + eps);
    }
    __syncthreads();
    float inv = bc;
    for (int i = threadIdx.x; i < D; i += 256) {
        float v = x[i] * inv;
        if (relu) v = fmaxf(v, 0.0f);
        if (of) of[row * (size_t)D + i] = v;
        if (ob) ob[row * (size_t)D + i] = to_bf16(v);
    }
}

// ---------------- per-(s,h,b) head squared-norms from bf16 tensor ----------------
__global__ __launch_bounds__(256) void headnorm_kernel(const __bf16* __restrict__ x,
                                                       float* __restrict__ hn) {
    int idx = blockIdx.x * blockDim.x + threadIdx.x;   // (s*NH + h)*BATCH + b
    if (idx >= S_LEN * NH * BATCH) return;
    int b = idx % BATCH;
    int h = (idx / BATCH) % NH;
    int s = idx / (BATCH * NH);
    const __bf16* p = x + ((size_t)s * BATCH + b) * DM + h * HD;
    float sum = 0.0f;
    for (int d = 0; d < HD; ++d) { float v = (float)p[d]; sum += v * v; }
    hn[idx] = sum;
}

// ---------------- UHG attention: one block per (16-query tile, head, batch) ----------------
__global__ __launch_bounds__(256) void attn_kernel(const __bf16* __restrict__ qb,
                                                   const __bf16* __restrict__ kb,
                                                   const __bf16* __restrict__ vb,
                                                   const float*  __restrict__ hnq,
                                                   const float*  __restrict__ hnk,
                                                   __bf16* __restrict__ outb) {
    __shared__ float  sc[16 * S_LEN];   // 128 KB: score tile (16 queries x full S)
    __shared__ __bf16 pr[16 * S_LEN];   //  64 KB: softmax probs (bf16)

    int lane = threadIdx.x & 31;
    int wave = threadIdx.x >> 5;
    int i0 = blockIdx.x * 16;
    int h  = blockIdx.y;
    int b  = blockIdx.z;
    size_t rstride = (size_t)BATCH * DM;

    // Q fragments for this 16-row tile (k-dim = head dim 64, two 32-steps)
    v16bf aq[2];
    {
        int m = i0 + (lane & 15);
        const __bf16* qrow = qb + (size_t)m * rstride + (size_t)b * DM + h * HD;
        #pragma unroll
        for (int kk = 0; kk < 2; ++kk) {
            const __bf16* p = qrow + kk * 32 + ((lane & 16) ? 16 : 0);
            #pragma unroll
            for (int t = 0; t < 8; ++t) { aq[kk][t] = p[t]; aq[kk][8 + t] = p[8 + t]; }
        }
    }

    // phase 1: dots -> UHG scores into LDS
    int jl = lane & 15, mo = (lane & 16) ? 8 : 0;
    for (int jt = wave; jt < S_LEN / 16; jt += 8) {
        v8f acc = {};
        #pragma unroll
        for (int kk = 0; kk < 2; ++kk) {
            int j = jt * 16 + jl;
            const __bf16* p = kb + (size_t)j * rstride + (size_t)b * DM + h * HD
                              + kk * 32 + ((lane & 16) ? 16 : 0);
            v16bf bf;
            #pragma unroll
            for (int u = 0; u < 16; ++u) bf[u] = p[u];
            acc = wmma_bf16(aq[kk], bf, acc);
        }
        float knj = hnk[((size_t)(jt * 16 + jl) * NH + h) * BATCH + b];
        #pragma unroll
        for (int r = 0; r < 8; ++r) {
            int m = mo + r;
            float qni = hnq[((size_t)(i0 + m) * NH + h) * BATCH + b];
            float d = acc[r];
            float quad = 1.0f - d * d / (qni * knj + 1e-8f);
            quad = fmaxf(quad, 0.0f);
            sc[m * S_LEN + jt * 16 + jl] = -sqrtf(quad);
        }
    }
    __syncthreads();

    // phase 2: softmax per row (2 rows per wave)
    #pragma unroll
    for (int rr = 0; rr < 2; ++rr) {
        int m = wave * 2 + rr;
        float mx = -3.0e38f;
        for (int j = lane; j < S_LEN; j += 32) mx = fmaxf(mx, sc[m * S_LEN + j]);
        #pragma unroll
        for (int o = 16; o; o >>= 1) mx = fmaxf(mx, __shfl_xor(mx, o, 32));
        float sum = 0.0f;
        for (int j = lane; j < S_LEN; j += 32) sum += __expf(sc[m * S_LEN + j] - mx);
        #pragma unroll
        for (int o = 16; o; o >>= 1) sum += __shfl_xor(sum, o, 32);
        float inv = 1.0f / sum;
        for (int j = lane; j < S_LEN; j += 32)
            pr[m * S_LEN + j] = to_bf16(__expf(sc[m * S_LEN + j] - mx) * inv);
    }
    __syncthreads();

    // phase 3: out(16 x 64) = P(16 x S) @ V(S x 64); waves 0..3 each own 16 head-dims
    if (wave < 4) {
        v8f acc = {};
        for (int k0 = 0; k0 < S_LEN; k0 += 32) {
            v16bf ap;
            {
                int m = lane & 15;
                const __bf16* p = &pr[m * S_LEN + k0 + ((lane & 16) ? 16 : 0)];
                #pragma unroll
                for (int t = 0; t < 8; ++t) { ap[t] = p[t]; ap[8 + t] = p[8 + t]; }
            }
            v16bf bv;
            {
                int kb2 = k0 + ((lane & 16) ? 16 : 0);
                int d   = h * HD + wave * 16 + jl;
                #pragma unroll
                for (int u = 0; u < 16; ++u)
                    bv[u] = vb[(size_t)(kb2 + u) * rstride + (size_t)b * DM + d];
            }
            acc = wmma_bf16(ap, bv, acc);
        }
        #pragma unroll
        for (int r = 0; r < 8; ++r) {
            int s = i0 + mo + r;
            outb[(size_t)s * rstride + (size_t)b * DM + h * HD + wave * 16 + jl] = to_bf16(acc[r]);
        }
    }
}

// ---------------- residual add + LayerNorm (ddof=1), D = 1024 ----------------
__global__ __launch_bounds__(256) void add_ln_kernel(const float* __restrict__ a,
                                                     const float* __restrict__ r,
                                                     const float* __restrict__ g,
                                                     const float* __restrict__ be,
                                                     float* __restrict__ of,
                                                     __bf16* __restrict__ ob) {
    size_t row = blockIdx.x;
    const float* pa = a + row * DM;
    const float* pr = r + row * DM;
    float xv[4];
    #pragma unroll
    for (int t = 0; t < 4; ++t) {
        int i = threadIdx.x + t * 256;
        xv[t] = pa[i] + pr[i];
    }
    __shared__ float red[8];
    __shared__ float bc;
    float s = xv[0] + xv[1] + xv[2] + xv[3];
    #pragma unroll
    for (int o = 16; o; o >>= 1) s += __shfl_xor(s, o, 32);
    if ((threadIdx.x & 31) == 0) red[threadIdx.x >> 5] = s;
    __syncthreads();
    if (threadIdx.x == 0) {
        float t = 0.0f;
        #pragma unroll
        for (int i = 0; i < 8; ++i) t += red[i];
        bc = t / (float)DM;
    }
    __syncthreads();
    float mean = bc;
    float s2 = 0.0f;
    #pragma unroll
    for (int t = 0; t < 4; ++t) { float d = xv[t] - mean; s2 += d * d; }
    #pragma unroll
    for (int o = 16; o; o >>= 1) s2 += __shfl_xor(s2, o, 32);
    __syncthreads();
    if ((threadIdx.x & 31) == 0) red[threadIdx.x >> 5] = s2;
    __syncthreads();
    if (threadIdx.x == 0) {
        float t = 0.0f;
        #pragma unroll
        for (int i = 0; i < 8; ++i) t += red[i];
        bc = rsqrtf(t / (float)(DM - 1) + 1e-5f);
    }
    __syncthreads();
    float inv = bc;
    #pragma unroll
    for (int t = 0; t < 4; ++t) {
        int i = threadIdx.x + t * 256;
        float o = (xv[t] - mean) * inv * g[i] + be[i];
        of[row * DM + i] = o;
        if (ob) ob[row * DM + i] = to_bf16(o);
    }
}

// ================================================================================
extern "C" void kernel_launch(void* const* d_in, const int* in_sizes, int n_in,
                              void* d_out, int out_size, void* d_ws, size_t ws_size,
                              hipStream_t stream) {
    (void)in_sizes; (void)n_in; (void)out_size; (void)ws_size;
    const float* src = (const float*)d_in[0];
    const float* wq = (const float*)d_in[1];  const float* bq = (const float*)d_in[2];
    const float* wk = (const float*)d_in[3];  const float* bk = (const float*)d_in[4];
    const float* wv = (const float*)d_in[5];  const float* bv = (const float*)d_in[6];
    const float* wo = (const float*)d_in[7];  const float* bo = (const float*)d_in[8];
    const float* w1 = (const float*)d_in[9];  const float* b1 = (const float*)d_in[10];
    const float* w2 = (const float*)d_in[11]; const float* b2 = (const float*)d_in[12];
    const float* g1 = (const float*)d_in[13]; const float* be1 = (const float*)d_in[14];
    const float* g2 = (const float*)d_in[15]; const float* be2 = (const float*)d_in[16];

    char* ws = (char*)d_ws;
    size_t off = 0;
    auto alloc = [&](size_t bytes) -> void* {
        void* p = ws + off;
        off += (bytes + 255) & ~(size_t)255;
        return p;
    };
    __bf16* wq_bf  = (__bf16*)alloc((size_t)DM * DM * 2);
    __bf16* wk_bf  = (__bf16*)alloc((size_t)DM * DM * 2);
    __bf16* wv_bf  = (__bf16*)alloc((size_t)DM * DM * 2);
    __bf16* wo_bf  = (__bf16*)alloc((size_t)DM * DM * 2);
    __bf16* w1_bf  = (__bf16*)alloc((size_t)DFF * DM * 2);
    __bf16* w2_bf  = (__bf16*)alloc((size_t)DM * DFF * 2);
    __bf16* src_bf = (__bf16*)alloc((size_t)MROWS * DM * 2);
    float*  gout   = (float*) alloc((size_t)MROWS * DFF * 4);   // shared GEMM output
    __bf16* q_bf   = (__bf16*)alloc((size_t)MROWS * DM * 2);
    __bf16* k_bf   = (__bf16*)alloc((size_t)MROWS * DM * 2);
    __bf16* v_bf   = (__bf16*)alloc((size_t)MROWS * DM * 2);
    float*  hnq    = (float*) alloc((size_t)S_LEN * NH * BATCH * 4);
    float*  hnk    = (float*) alloc((size_t)S_LEN * NH * BATCH * 4);
    __bf16* attn_bf  = (__bf16*)alloc((size_t)MROWS * DM * 2);
    float*  attnproj = (float*) alloc((size_t)MROWS * DM * 4);
    float*  x_f    = (float*) alloc((size_t)MROWS * DM * 4);
    __bf16* x_bf   = (__bf16*)alloc((size_t)MROWS * DM * 2);
    __bf16* h_bf   = (__bf16*)alloc((size_t)MROWS * DFF * 2);
    float*  ff_f   = (float*) alloc((size_t)MROWS * DM * 4);
    float*  y_f    = (float*) alloc((size_t)MROWS * DM * 4);

    auto cvt = [&](const float* in, __bf16* out, size_t n) {
        cvt_kernel<<<(unsigned)((n + 255) / 256), 256, 0, stream>>>(in, out, n);
    };
    cvt(wq, wq_bf, (size_t)DM * DM);
    cvt(wk, wk_bf, (size_t)DM * DM);
    cvt(wv, wv_bf, (size_t)DM * DM);
    cvt(wo, wo_bf, (size_t)DM * DM);
    cvt(w1, w1_bf, (size_t)DFF * DM);
    cvt(w2, w2_bf, (size_t)DM * DFF);
    cvt(src, src_bf, (size_t)MROWS * DM);

    auto gemm = [&](const __bf16* A, const __bf16* W, const float* bias, float* C,
                    int M, int N, int K) {
        dim3 grid(N / BN, M / BM);
        gemm_bias_kernel<<<grid, 256, 0, stream>>>(A, W, bias, C, M, N, K);
    };

    // Q / K / V projections + unit-sphere projection (bf16 outputs for WMMA)
    gemm(src_bf, wq_bf, bq, gout, MROWS, DM, DM);
    rownorm_kernel<<<MROWS, 256, 0, stream>>>(gout, nullptr, q_bf, DM, 1e-8f, 0);
    gemm(src_bf, wk_bf, bk, gout, MROWS, DM, DM);
    rownorm_kernel<<<MROWS, 256, 0, stream>>>(gout, nullptr, k_bf, DM, 1e-8f, 0);
    gemm(src_bf, wv_bf, bv, gout, MROWS, DM, DM);
    rownorm_kernel<<<MROWS, 256, 0, stream>>>(gout, nullptr, v_bf, DM, 1e-8f, 0);

    // per-head squared norms
    headnorm_kernel<<<(S_LEN * NH * BATCH) / 256, 256, 0, stream>>>(q_bf, hnq);
    headnorm_kernel<<<(S_LEN * NH * BATCH) / 256, 256, 0, stream>>>(k_bf, hnk);

    // UHG attention
    {
        dim3 grid(S_LEN / 16, NH, BATCH);
        attn_kernel<<<grid, 256, 0, stream>>>(q_bf, k_bf, v_bf, hnq, hnk, attn_bf);
    }

    // output projection + project
    gemm(attn_bf, wo_bf, bo, gout, MROWS, DM, DM);
    rownorm_kernel<<<MROWS, 256, 0, stream>>>(gout, attnproj, nullptr, DM, 1e-8f, 0);

    // x = LN(src + attn_out)
    add_ln_kernel<<<MROWS, 256, 0, stream>>>(src, attnproj, g1, be1, x_f, x_bf);

    // FFN: h = relu(project(x @ w1^T + b1))
    gemm(x_bf, w1_bf, b1, gout, MROWS, DFF, DM);
    rownorm_kernel<<<MROWS, 256, 0, stream>>>(gout, nullptr, h_bf, DFF, 1e-8f, 1);

    // ff = project(h @ w2^T + b2)
    gemm(h_bf, w2_bf, b2, gout, MROWS, DM, DFF);
    rownorm_kernel<<<MROWS, 256, 0, stream>>>(gout, ff_f, nullptr, DM, 1e-8f, 0);

    // x = LN(x + ff); final exact unit normalize (no eps)
    add_ln_kernel<<<MROWS, 256, 0, stream>>>(x_f, ff_f, g2, be2, y_f, nullptr);
    rownorm_kernel<<<MROWS, 256, 0, stream>>>(y_f, (float*)d_out, nullptr, DM, 0.0f, 0);
}
